// MiniAttention_6012954214731
// MI455X (gfx1250) — matrix operand order repaired and checked by reading, not compiled, not run on hardware
//
#include <hip/hip_runtime.h>
#include <hip/hip_bf16.h>

// MiniAttention for MI455X (gfx1250, wave32, WMMA).
// Pipeline:
//  1) cvt: f32 -> bf16 for x, w_qkv, w_proj
//  2) qkv_gemm (WMMA bf16): qkv = x @ w_qkv^T, epilogue scatters to
//     Q[b,h,n,d]*SCALE, K[b,h,n,d], V^T[b,h,d,n]  (all bf16)
//  3) per batch b:
//     a) s_gemm (WMMA, 32x64 wave tiles): S[h,n,m] = Q_h K_h^T  (f32)
//     b) fused_mix_softmax: mix1(conv_l) -> softmax(m) -> mix2(conv_w),
//        LDS filled with GLOBAL_LOAD_ASYNC_TO_LDS_B128 (ASYNCcnt path),
//        writes P[h,n,m] bf16
//     c) av_gemm (WMMA): out_pre[n, h*64+d] = P_h @ V_h
//  4) proj_gemm (WMMA): out = out_pre @ w_proj^T + b_proj  (f32)

#define Bc 8
#define Nc 1024
#define Cc 768
#define Hc 12
#define Dc 64

typedef __attribute__((ext_vector_type(16))) __bf16 bf16x16;
typedef __attribute__((ext_vector_type(8)))  __bf16 bf16x8;
typedef __attribute__((ext_vector_type(8)))  float  f32x8;

__device__ __forceinline__ __bf16 f2bf(float f) {
  unsigned u = __float_as_uint(f);
  unsigned r = u + 0x7FFFu + ((u >> 16) & 1u);   // round-to-nearest-even
  unsigned short h = (unsigned short)(r >> 16);
  return __builtin_bit_cast(__bf16, h);
}

// A-fragment (16x32 bf16): lane l=lane&15 -> row M=l; hi=lane>>4.
// VGPR0-7: K = hi*16+0..15  (one contiguous run)
__device__ __forceinline__ bf16x16 load_a_frag(const __bf16* rowbase, size_t ld,
                                               int row0, int k0, int lane) {
  int l = lane & 15, hi = lane >> 4;
  const __bf16* p = rowbase + (size_t)(row0 + l) * ld + k0 + hi * 16;
  bf16x8 lo = *(const bf16x8*)p;
  bf16x8 hh = *(const bf16x8*)(p + 8);
  bf16x16 a;
#pragma unroll
  for (int t = 0; t < 8; ++t) { a[t] = lo[t]; a[t + 8] = hh[t]; }
  return a;
}

// B-fragment (32x16 bf16), column-major in memory (= row-major [n][k]):
// lane l -> column n=col0+l ; K run = k0 + hi*16, 16 contiguous elements.
__device__ __forceinline__ bf16x16 load_b_frag(const __bf16* colbase, size_t ld,
                                               int col0, int k0, int lane) {
  int l = lane & 15, hi = lane >> 4;
  return *(const bf16x16*)(colbase + (size_t)(col0 + l) * ld + k0 + hi * 16);
}

__global__ __launch_bounds__(256) void cvt_bf16(const float* __restrict__ src,
                                                __bf16* __restrict__ dst, int n) {
  int i = blockIdx.x * 256 + threadIdx.x;
  if (i < n) dst[i] = f2bf(src[i]);
}

// ---------------- QKV GEMM: M=8192, N=2304, K=768, wave tile 32x64 ----------
__global__ __launch_bounds__(256) void qkv_gemm(const __bf16* __restrict__ xb,
                                                const __bf16* __restrict__ wb,
                                                __bf16* __restrict__ Qb,
                                                __bf16* __restrict__ Kb,
                                                __bf16* __restrict__ Vtb) {
  int warp = threadIdx.x >> 5, lane = threadIdx.x & 31;
  int wt = blockIdx.x * 8 + warp;               // 256*36 = 9216 wave tiles
  int mt = wt / 36, ntile = wt % 36;
  int m0 = mt * 32, n0 = ntile * 64;
  int l = lane & 15, hi = lane >> 4;

  f32x8 acc[2][4] = {};
  for (int k0 = 0; k0 < Cc; k0 += 32) {
    bf16x16 a[2], bf[4];
#pragma unroll
    for (int i = 0; i < 2; ++i) a[i] = load_a_frag(xb, Cc, m0 + 16 * i, k0, lane);
#pragma unroll
    for (int j = 0; j < 4; ++j) bf[j] = load_b_frag(wb, Cc, n0 + 16 * j, k0, lane);
#pragma unroll
    for (int i = 0; i < 2; ++i)
#pragma unroll
      for (int j = 0; j < 4; ++j)
        acc[i][j] = __builtin_amdgcn_wmma_f32_16x16x32_bf16(
            false, a[i], false, bf[j], (short)0, acc[i][j], false, false);
  }
#pragma unroll
  for (int i = 0; i < 2; ++i)
#pragma unroll
    for (int j = 0; j < 4; ++j) {
      int col = n0 + 16 * j + l;
      int which = col / Cc, rem = col % Cc;
      int h = rem >> 6, d = rem & 63;
#pragma unroll
      for (int r = 0; r < 8; ++r) {
        int row = m0 + 16 * i + r + hi * 8;     // token index b*N+n
        int bb = row >> 10, nn = row & 1023;
        float v = acc[i][j][r];
        if (which == 0)
          Qb[(((size_t)bb * Hc + h) * Nc + nn) * Dc + d] = f2bf(v * 0.125f);
        else if (which == 1)
          Kb[(((size_t)bb * Hc + h) * Nc + nn) * Dc + d] = f2bf(v);
        else
          Vtb[(((size_t)bb * Hc + h) * Dc + d) * Nc + nn] = f2bf(v);
      }
    }
}

// ---------------- S = Q K^T per head, wave tile 32(n) x 64(m) ---------------
__global__ __launch_bounds__(256) void s_gemm(const __bf16* __restrict__ Qb,
                                              const __bf16* __restrict__ Kb,
                                              float* __restrict__ attn1, int b) {
  int warp = threadIdx.x >> 5, lane = threadIdx.x & 31;
  int wt = blockIdx.x * 8 + warp;               // 12*32*16 = 6144 wave tiles
  int h = wt / 512, rem = wt % 512;
  int nt = rem >> 4, mt = rem & 15;             // nt: 32-row tile, mt: 64-col tile
  int l = lane & 15, hi = lane >> 4;
  const __bf16* Qh = Qb + ((size_t)b * Hc + h) * Nc * Dc;
  const __bf16* Kh = Kb + ((size_t)b * Hc + h) * Nc * Dc;

  f32x8 acc[2][4] = {};
#pragma unroll
  for (int k0 = 0; k0 < Dc; k0 += 32) {
    bf16x16 a[2];
#pragma unroll
    for (int i = 0; i < 2; ++i) a[i] = load_a_frag(Qh, Dc, nt * 32 + 16 * i, k0, lane);
#pragma unroll
    for (int j = 0; j < 4; ++j) {
      bf16x16 bfr = load_b_frag(Kh, Dc, mt * 64 + 16 * j, k0, lane);
#pragma unroll
      for (int i = 0; i < 2; ++i)
        acc[i][j] = __builtin_amdgcn_wmma_f32_16x16x32_bf16(
            false, a[i], false, bfr, (short)0, acc[i][j], false, false);
    }
  }
  float* oh = attn1 + ((size_t)h << 20);
#pragma unroll
  for (int i = 0; i < 2; ++i)
#pragma unroll
    for (int j = 0; j < 4; ++j)
#pragma unroll
      for (int r = 0; r < 8; ++r) {
        int n = nt * 32 + 16 * i + r + hi * 8;
        int m = mt * 64 + 16 * j + l;
        oh[((size_t)n << 10) + m] = acc[i][j][r];
      }
}

// -------- fused: mix1(conv_l) -> softmax over m -> mix2(conv_w) -------------
// one block per query row n; all 12 heads resident in LDS.
// LDS fill uses CDNA5 GLOBAL_LOAD_ASYNC_TO_LDS_B128 (ASYNCcnt).
__global__ __launch_bounds__(256) void fused_mix_softmax(
    const float* __restrict__ attn1, const float* __restrict__ convl,
    const float* __restrict__ convw, __bf16* __restrict__ attnP) {
  __shared__ float ls[Hc][Nc];                  // 48 KB
  __shared__ float cl[Hc * Hc], cw[Hc * Hc];
  __shared__ float red[Hc][8];
  int tid = threadIdx.x, n = blockIdx.x;
  int lane = tid & 31, w = tid >> 5;

  if (tid < Hc * Hc) { cl[tid] = convl[tid]; cw[tid] = convw[tid]; }
  // async DMA fill: 12*1024 floats = 3072 x 16B chunks, 12 per thread.
  // low 32 bits of the flat shared address = wave-relative LDS byte offset.
  unsigned lsbase = (unsigned)(size_t)(&ls[0][0]);
#pragma unroll
  for (int it = 0; it < 12; ++it) {
    int c = tid + it * 256;                     // chunk id 0..3071
    int h = c >> 8, rr = c & 255;               // 256 x 16B chunks per head row
    const float* g = attn1 + ((size_t)h << 20) + ((size_t)n << 10) + rr * 4;
    unsigned ldsa = lsbase + (unsigned)(h * (Nc * 4) + rr * 16);
    asm volatile("global_load_async_to_lds_b128 %0, %1, off"
                 :: "v"(ldsa), "v"(g) : "memory");
  }
  asm volatile("s_wait_asynccnt 0" ::: "memory");
  __syncthreads();

  float ev[4][Hc], tmax[Hc];
#pragma unroll
  for (int o = 0; o < Hc; ++o) tmax[o] = -3.4e38f;
#pragma unroll
  for (int j = 0; j < 4; ++j) {
    int m = tid + j * 256;
    float s[Hc];
#pragma unroll
    for (int h = 0; h < Hc; ++h) s[h] = ls[h][m];
#pragma unroll
    for (int o = 0; o < Hc; ++o) {
      float acc = 0.f;
#pragma unroll
      for (int h = 0; h < Hc; ++h) acc = fmaf(cl[o * Hc + h], s[h], acc);
      ev[j][o] = acc;
      tmax[o] = fmaxf(tmax[o], acc);
    }
  }
  // block-reduce max per head
#pragma unroll
  for (int o = 0; o < Hc; ++o) {
    float v = tmax[o];
#pragma unroll
    for (int off = 16; off > 0; off >>= 1) v = fmaxf(v, __shfl_xor(v, off, 32));
    if (lane == 0) red[o][w] = v;
  }
  __syncthreads();
  float rmax[Hc];
#pragma unroll
  for (int o = 0; o < Hc; ++o) {
    float v = red[o][0];
#pragma unroll
    for (int i = 1; i < 8; ++i) v = fmaxf(v, red[o][i]);
    rmax[o] = v;
  }
  __syncthreads();

  float tsum[Hc];
#pragma unroll
  for (int o = 0; o < Hc; ++o) tsum[o] = 0.f;
#pragma unroll
  for (int j = 0; j < 4; ++j)
#pragma unroll
    for (int o = 0; o < Hc; ++o) {
      float e = __expf(ev[j][o] - rmax[o]);
      ev[j][o] = e;
      tsum[o] += e;
    }
#pragma unroll
  for (int o = 0; o < Hc; ++o) {
    float v = tsum[o];
#pragma unroll
    for (int off = 16; off > 0; off >>= 1) v += __shfl_xor(v, off, 32);
    if (lane == 0) red[o][w] = v;
  }
  __syncthreads();
  float rinv[Hc];
#pragma unroll
  for (int o = 0; o < Hc; ++o) {
    float v = red[o][0];
#pragma unroll
    for (int i = 1; i < 8; ++i) v += red[o][i];
    rinv[o] = 1.0f / v;
  }

#pragma unroll
  for (int j = 0; j < 4; ++j) {
    int m = tid + j * 256;
    float p[Hc];
#pragma unroll
    for (int h = 0; h < Hc; ++h) p[h] = ev[j][h] * rinv[h];
#pragma unroll
    for (int o = 0; o < Hc; ++o) {
      float acc = 0.f;
#pragma unroll
      for (int h = 0; h < Hc; ++h) acc = fmaf(cw[o * Hc + h], p[h], acc);
      attnP[((size_t)o << 20) + ((size_t)n << 10) + m] = f2bf(acc);
    }
  }
}

// ---------------- out_pre = P @ V per head, wave tile 16(n) x 64(d) ---------
__global__ __launch_bounds__(256) void av_gemm(const __bf16* __restrict__ attnP,
                                               const __bf16* __restrict__ Vtb,
                                               __bf16* __restrict__ outpre, int b) {
  int warp = threadIdx.x >> 5, lane = threadIdx.x & 31;
  int wt = blockIdx.x * 8 + warp;               // 12*64 = 768 wave tiles
  int h = wt >> 6, nt = wt & 63;
  int l = lane & 15, hi = lane >> 4;
  const __bf16* Ph = attnP + ((size_t)h << 20);             // [n][m], m contig
  const __bf16* Vh = Vtb + ((size_t)b * Hc + h) * Dc * Nc;  // [d][m], m contig

  f32x8 acc[4] = {};
  for (int k0 = 0; k0 < Nc; k0 += 32) {
    bf16x16 a = load_a_frag(Ph, Nc, nt * 16, k0, lane);
#pragma unroll
    for (int j = 0; j < 4; ++j) {
      bf16x16 bfr = load_b_frag(Vh, Nc, 16 * j, k0, lane);
      acc[j] = __builtin_amdgcn_wmma_f32_16x16x32_bf16(
          false, a, false, bfr, (short)0, acc[j], false, false);
    }
  }
#pragma unroll
  for (int j = 0; j < 4; ++j)
#pragma unroll
    for (int r = 0; r < 8; ++r) {
      int n = nt * 16 + r + hi * 8;
      int d = 16 * j + l;
      outpre[((size_t)(b * Nc + n)) * Cc + h * Dc + d] = f2bf(acc[j][r]);
    }
}

// ---------------- final projection: M=8192, N=768, K=768 --------------------
__global__ __launch_bounds__(256) void proj_gemm(const __bf16* __restrict__ op,
                                                 const __bf16* __restrict__ wpb,
                                                 const float* __restrict__ bias,
                                                 float* __restrict__ out) {
  int warp = threadIdx.x >> 5, lane = threadIdx.x & 31;
  int wt = blockIdx.x * 8 + warp;               // 256*12 = 3072 wave tiles
  int mt = wt / 12, ntile = wt % 12;
  int m0 = mt * 32, n0 = ntile * 64;
  int l = lane & 15, hi = lane >> 4;

  f32x8 acc[2][4] = {};
  for (int k0 = 0; k0 < Cc; k0 += 32) {
    bf16x16 a[2], bf[4];
#pragma unroll
    for (int i = 0; i < 2; ++i) a[i] = load_a_frag(op, Cc, m0 + 16 * i, k0, lane);
#pragma unroll
    for (int j = 0; j < 4; ++j) bf[j] = load_b_frag(wpb, Cc, n0 + 16 * j, k0, lane);
#pragma unroll
    for (int i = 0; i < 2; ++i)
#pragma unroll
      for (int j = 0; j < 4; ++j)
        acc[i][j] = __builtin_amdgcn_wmma_f32_16x16x32_bf16(
            false, a[i], false, bf[j], (short)0, acc[i][j], false, false);
  }
#pragma unroll
  for (int i = 0; i < 2; ++i)
#pragma unroll
    for (int j = 0; j < 4; ++j) {
      int col = n0 + 16 * j + l;
      float bv = bias[col];
#pragma unroll
      for (int r = 0; r < 8; ++r) {
        int row = m0 + 16 * i + r + hi * 8;
        out[(size_t)row * Cc + col] = acc[i][j][r] + bv;
      }
    }
}

extern "C" void kernel_launch(void* const* d_in, const int* in_sizes, int n_in,
                              void* d_out, int out_size, void* d_ws, size_t ws_size,
                              hipStream_t stream) {
  (void)in_sizes; (void)n_in; (void)out_size; (void)ws_size;
  const float* x      = (const float*)d_in[0];
  const float* w_qkv  = (const float*)d_in[1];
  const float* w_proj = (const float*)d_in[2];
  const float* b_proj = (const float*)d_in[3];
  const float* conv_l = (const float*)d_in[4];
  const float* conv_w = (const float*)d_in[5];
  float* out = (float*)d_out;

  size_t off = 0;
  auto alloc = [&](size_t bytes) -> char* {
    char* p = (char*)d_ws + off;
    off += (bytes + 255) & ~(size_t)255;
    return p;
  };
  const size_t n_x   = (size_t)Bc * Nc * Cc;          // 6.29M
  const size_t n_wq  = (size_t)3 * Cc * Cc;           // 1.77M
  const size_t n_wp  = (size_t)Cc * Cc;               // 0.59M
  const size_t n_hnd = (size_t)Bc * Hc * Nc * Dc;     // 6.29M

  __bf16* xb     = (__bf16*)alloc(n_x   * 2);
  __bf16* wqkvb  = (__bf16*)alloc(n_wq  * 2);
  __bf16* wprojb = (__bf16*)alloc(n_wp  * 2);
  __bf16* Qb     = (__bf16*)alloc(n_hnd * 2);
  __bf16* Kb     = (__bf16*)alloc(n_hnd * 2);
  __bf16* Vtb    = (__bf16*)alloc(n_hnd * 2);
  __bf16* outpre = (__bf16*)alloc(n_x   * 2);
  float*  attn1  = (float*) alloc((size_t)Hc * Nc * Nc * 4);   // 50 MB, reused per b
  __bf16* attnP  = (__bf16*)alloc((size_t)Hc * Nc * Nc * 2);   // 25 MB, reused per b

  // 1) convert to bf16
  cvt_bf16<<<(int)((n_x  + 255) / 256), 256, 0, stream>>>(x,      xb,     (int)n_x);
  cvt_bf16<<<(int)((n_wq + 255) / 256), 256, 0, stream>>>(w_qkv,  wqkvb,  (int)n_wq);
  cvt_bf16<<<(int)((n_wp + 255) / 256), 256, 0, stream>>>(w_proj, wprojb, (int)n_wp);

  // 2) QKV projection: 9216 wave tiles / 8 waves per block
  qkv_gemm<<<1152, 256, 0, stream>>>(xb, wqkvb, Qb, Kb, Vtb);

  // 3) per-batch attention (buffers reused; stream order serializes)
  for (int b = 0; b < Bc; ++b) {
    s_gemm<<<768, 256, 0, stream>>>(Qb, Kb, attn1, b);
    fused_mix_softmax<<<Nc, 256, 0, stream>>>(attn1, conv_l, conv_w, attnP);
    av_gemm<<<96, 256, 0, stream>>>(attnP, Vtb, outpre, b);
  }

  // 4) output projection
  proj_gemm<<<384, 256, 0, stream>>>(outpre, wprojb, b_proj, out);
}
